// StaticGNN_Identity_Entropy_53171695124551
// MI455X (gfx1250) — matrix operand order repaired and checked by reading, not compiled, not run on hardware
//
#include <hip/hip_runtime.h>
#include <cstdint>
#include <cstddef>

// ---------------------------------------------------------------------------
// Problem constants (from reference)
// ---------------------------------------------------------------------------
constexpr int kN   = 50000;        // nodes
constexpr int kE   = 400000;       // edges
constexpr int kND  = 128;          // node feature dim (edge-MLP out)
constexpr int kED  = 64;           // edge attr dim
constexpr int kHID = 128;          // hidden
constexpr int kGIN = 288;          // 2*ND+2 = 258 padded up to 32-multiple
constexpr int kCIN = 320;          // 2*HID + ED
constexpr int kE2  = kE + kN;      // edges + self loops

typedef __attribute__((ext_vector_type(16))) _Float16 v16h;
typedef __attribute__((ext_vector_type(8)))  _Float16 v8h;
typedef __attribute__((ext_vector_type(8)))  float    v8f;

#define WMMA_F16 __builtin_amdgcn_wmma_f32_16x16x32_f16

// ---------------------------------------------------------------------------
// WMMA fragment helpers (CDNA5 wave32 layouts, CDNA5 ISA §7.12.2)
// ---------------------------------------------------------------------------
// A: 16x32 f16 from row-major LDS tile; row M = lane&15.
// K map: lanes0-15 K={0..15}, lanes16-31 +16. Each 8-element half is
// contiguous in LDS -> compiler emits 2x ds_load_b128.
__device__ __forceinline__ v16h wmma_load_a_lds(const _Float16* As, int lda,
                                                int k0, int lane) {
  int m = lane & 15, hf = lane >> 4;
  v16h a;
#pragma unroll
  for (int i = 0; i < 16; ++i) {
    int k = i + 16 * hf;   // {0..15} + 16*half
    a[i] = As[m * lda + k0 + k];
  }
  return a;
}
// B: pre-swizzled weights. Fragment for (kb=k0/32, nb=n0/16) is stored as
// 32 lanes x 16 f16 contiguous; each lane reads 32 aligned bytes
// (-> 2x global_load_b128). Element (lane,i) == W[k0+i+16*(lane>>4)][n0+(lane&15)].
__device__ __forceinline__ v16h wmma_load_b_sw(const _Float16* Wsw, int ncols,
                                               int k0, int n0, int lane) {
  size_t idx = ((((size_t)(k0 >> 5) * (size_t)(ncols >> 4)) + (size_t)(n0 >> 4)) * 32
                + (size_t)lane) * 16;
  return *(const v16h*)(Wsw + idx);
}

// monotone float->uint key for atomicMax-based segment max
__device__ __forceinline__ unsigned fkey(float f) {
  unsigned u = __float_as_uint(f);
  return (u & 0x80000000u) ? ~u : (u | 0x80000000u);
}
__device__ __forceinline__ float funkey(unsigned u) {
  unsigned b = (u & 0x80000000u) ? (u & 0x7FFFFFFFu) : ~u;
  return __uint_as_float(b);
}

// ---------------------------------------------------------------------------
// Utility kernels
// ---------------------------------------------------------------------------
__global__ void k_cvt(const float* __restrict__ src, _Float16* __restrict__ dst,
                      int srcRows, int cols, long total) {
  long i = (long)blockIdx.x * blockDim.x + threadIdx.x;
  if (i >= total) return;
  long r = i / cols;
  int  c = (int)(i - r * cols);
  dst[i] = (_Float16)(r < srcRows ? src[r * (long)cols + c] : 0.f);
}

// f32 row-major weight [srcRows, ncols] -> f16 WMMA-fragment-swizzled buffer
// of ceil(srcRows/32)*32 x ncols elements (zero padded K rows).
__global__ void k_swz_b(const float* __restrict__ W, _Float16* __restrict__ Wsw,
                        int srcRows, int ncols, long total) {
  long o = (long)blockIdx.x * blockDim.x + threadIdx.x;
  if (o >= total) return;
  int i    = (int)(o & 15);
  int lane = (int)((o >> 4) & 31);
  long t   = o >> 9;
  int nbc  = ncols >> 4;
  int nb   = (int)(t % nbc);
  int kb   = (int)(t / nbc);
  int k = kb * 32 + i + 16 * (lane >> 4);
  int n = nb * 16 + (lane & 15);
  Wsw[o] = (_Float16)(k < srcRows ? W[(size_t)k * ncols + n] : 0.f);
}

__global__ void k_count(const int* __restrict__ src, const int* __restrict__ dst,
                        float* __restrict__ cnt_out, float* __restrict__ cnt_in) {
  int e = blockIdx.x * blockDim.x + threadIdx.x;
  if (e >= kE) return;
  atomicAdd(&cnt_out[src[e]], 1.f);
  atomicAdd(&cnt_in[dst[e]], 1.f);
}

__global__ void k_loopsum(const float* __restrict__ ea, const int* __restrict__ dst,
                          float* __restrict__ loop_sum) {
  long i = (long)blockIdx.x * blockDim.x + threadIdx.x;
  if (i >= (long)kE * kED) return;
  int e = (int)(i >> 6), c = (int)(i & 63);
  atomicAdd(&loop_sum[(size_t)dst[e] * kED + c], ea[i]);
}

// edge MLP:  emb = relu(edge_attr @ epw + epb), scattered into seg-sums
__global__ void k_edge_emb(const float* __restrict__ ea, const int* __restrict__ src,
                           const int* __restrict__ dst, const _Float16* __restrict__ epw_sw,
                           const float* __restrict__ epb, float* __restrict__ xsum_out,
                           float* __restrict__ xsum_in) {
  __shared__ _Float16 At[16 * kED];
  __shared__ int sIdx[16], dIdx[16];
  int tid = threadIdx.x, tile = blockIdx.x;
  for (int i = tid; i < 16 * kED; i += 256) {
    int m = i / kED, c = i % kED;
    At[i] = (_Float16)ea[(size_t)(tile * 16 + m) * kED + c];
  }
  if (tid < 16) { sIdx[tid] = src[tile * 16 + tid]; dIdx[tid] = dst[tile * 16 + tid]; }
  __syncthreads();
  int wave = tid >> 5, lane = tid & 31;
  v8f acc = {0.f, 0.f, 0.f, 0.f, 0.f, 0.f, 0.f, 0.f};
#pragma unroll
  for (int k0 = 0; k0 < kED; k0 += 32)
    acc = WMMA_F16(false, wmma_load_a_lds(At, kED, k0, lane),
                   false, wmma_load_b_sw(epw_sw, kND, k0, wave * 16, lane),
                   (short)0, acc, false, false);
  int col = wave * 16 + (lane & 15), hf = lane >> 4;
  float bias = epb[col];
#pragma unroll
  for (int r = 0; r < 8; ++r) {
    int m = r + 8 * hf;
    float v = fmaxf(acc[r] + bias, 0.f);
    atomicAdd(&xsum_out[(size_t)sIdx[m] * kND + col], v);
    atomicAdd(&xsum_in[(size_t)dIdx[m] * kND + col], v);
  }
}

// xi = [x_out | x_in | node_stats | 0-pad] as f16 [N, 288]
__global__ void k_build_xi(const float* __restrict__ xsum_out, const float* __restrict__ xsum_in,
                           const float* __restrict__ cnt_out, const float* __restrict__ cnt_in,
                           const float* __restrict__ stats, _Float16* __restrict__ xi) {
  long i = (long)blockIdx.x * blockDim.x + threadIdx.x;
  if (i >= (long)kN * kGIN) return;
  int n = (int)(i / kGIN), c = (int)(i % kGIN);
  float v;
  if (c < 128)       v = xsum_out[(size_t)n * kND + c] / fmaxf(cnt_out[n], 1.f);
  else if (c < 256)  v = xsum_in[(size_t)n * kND + (c - 128)] / fmaxf(cnt_in[n], 1.f);
  else if (c < 258)  v = stats[(size_t)n * 2 + (c - 256)];
  else               v = 0.f;
  xi[i] = (_Float16)v;
}

__global__ void k_finalize_loop(const float* __restrict__ loop_sum,
                                const float* __restrict__ cnt_in,
                                _Float16* __restrict__ loop16) {
  long i = (long)blockIdx.x * blockDim.x + threadIdx.x;
  if (i >= (long)kN * kED) return;
  int n = (int)(i >> 6);
  loop16[i] = (_Float16)(loop_sum[i] / fmaxf(cnt_in[n], 1.f));
}

// generic C[M,ncols](f16) = A[M,K](f16) @ Bsw[K,ncols](f16, swizzled);
// grid=(M/16, ncols/64), block=128
__global__ void k_gemm16(const _Float16* __restrict__ A, const _Float16* __restrict__ Bsw,
                         _Float16* __restrict__ C, int K, int ncols) {
  __shared__ _Float16 At[16 * 320];
  int tid = threadIdx.x, tile = blockIdx.x;
  for (int i = tid; i < 16 * K; i += 128)
    At[i] = A[(size_t)(tile * 16 + i / K) * K + i % K];
  __syncthreads();
  int wave = tid >> 5, lane = tid & 31;
  int n0 = blockIdx.y * 64 + wave * 16;
  v8f acc = {0.f, 0.f, 0.f, 0.f, 0.f, 0.f, 0.f, 0.f};
  for (int k0 = 0; k0 < K; k0 += 32)
    acc = WMMA_F16(false, wmma_load_a_lds(At, K, k0, lane),
                   false, wmma_load_b_sw(Bsw, ncols, k0, n0, lane),
                   (short)0, acc, false, false);
  int col = n0 + (lane & 15), hf = lane >> 4;
#pragma unroll
  for (int r = 0; r < 8; ++r)
    C[(size_t)(tile * 16 + r + 8 * hf) * ncols + col] = (_Float16)acc[r];
}

// GATv2 attention logits + segment-max over dst (v8h vectorized gathers)
__global__ void k_logit(const _Float16* __restrict__ xl, const _Float16* __restrict__ xr,
                        const _Float16* __restrict__ weE, const _Float16* __restrict__ weL,
                        const float* __restrict__ att, const int* __restrict__ src,
                        const int* __restrict__ dst, float* __restrict__ logit,
                        unsigned* __restrict__ mkey, int heads) {
  int idx = blockIdx.x * blockDim.x + threadIdx.x;
  if (idx >= kE2 * heads) return;
  int e2 = idx / heads, h = idx - e2 * heads;
  int ldx = heads * kHID;
  int s, d; const _Float16* we;
  if (e2 < kE) { s = src[e2]; d = dst[e2]; we = weE + (size_t)e2 * ldx + h * kHID; }
  else         { s = d = e2 - kE;          we = weL + (size_t)s  * ldx + h * kHID; }
  const v8h* xl8 = (const v8h*)(xl + (size_t)s * ldx + h * kHID);
  const v8h* xr8 = (const v8h*)(xr + (size_t)d * ldx + h * kHID);
  const v8h* we8 = (const v8h*)we;
  const float* ap = att + h * kHID;
  __builtin_prefetch(xl8, 0, 0);
  __builtin_prefetch(xr8, 0, 0);
  float acc = 0.f;
#pragma unroll 4
  for (int cc = 0; cc < kHID / 8; ++cc) {
    v8h a = xl8[cc], b = xr8[cc], w = we8[cc];
#pragma unroll
    for (int j = 0; j < 8; ++j) {
      float v = (float)a[j] + (float)b[j] + (float)w[j];
      v = v > 0.f ? v : 0.2f * v;               // leaky_relu 0.2
      acc += v * ap[cc * 8 + j];
    }
  }
  logit[idx] = acc;
  atomicMax(&mkey[d * heads + h], fkey(acc));
}

__global__ void k_denom(const float* __restrict__ logit, const unsigned* __restrict__ mkey,
                        float* __restrict__ denom, const int* __restrict__ dst, int heads) {
  int idx = blockIdx.x * blockDim.x + threadIdx.x;
  if (idx >= kE2 * heads) return;
  int e2 = idx / heads, h = idx - e2 * heads;
  int d = (e2 < kE) ? dst[e2] : e2 - kE;
  float a = __expf(logit[idx] - funkey(mkey[d * heads + h]));
  atomicAdd(&denom[d * heads + h], a);
}

__global__ void k_attn_accum(const _Float16* __restrict__ xl, const float* __restrict__ logit,
                             const unsigned* __restrict__ mkey, const float* __restrict__ denom,
                             const int* __restrict__ src, const int* __restrict__ dst,
                             float* __restrict__ out, int heads) {
  int idx = blockIdx.x * blockDim.x + threadIdx.x;
  if (idx >= kE2 * heads) return;
  int e2 = idx / heads, h = idx - e2 * heads;
  int ldx = heads * kHID;
  int s, d;
  if (e2 < kE) { s = src[e2]; d = dst[e2]; } else { s = d = e2 - kE; }
  const v8h* xl8 = (const v8h*)(xl + (size_t)s * ldx + h * kHID);
  __builtin_prefetch(xl8, 0, 0);
  float a = __expf(logit[idx] - funkey(mkey[d * heads + h])) /
            fmaxf(denom[d * heads + h], 1e-16f);
  float* op = out + (size_t)d * ldx + h * kHID;
#pragma unroll 2
  for (int cc = 0; cc < kHID / 8; ++cc) {
    v8h x = xl8[cc];
#pragma unroll
    for (int j = 0; j < 8; ++j) atomicAdd(&op[cc * 8 + j], (float)x[j] * a);
  }
}

// mean over heads + bias -> layernorm -> elu ; one wave per node
__global__ void k_node_norm(const float* __restrict__ accum, int heads,
                            const float* __restrict__ bias, const float* __restrict__ gam,
                            const float* __restrict__ bet, _Float16* __restrict__ hout) {
  int wave = threadIdx.x >> 5, lane = threadIdx.x & 31;
  int node = blockIdx.x * 8 + wave;
  if (node >= kN) return;
  const float inv_h = 1.f / (float)heads;
  float v[4], s1 = 0.f, s2 = 0.f;
#pragma unroll
  for (int j = 0; j < 4; ++j) {
    int c = lane + 32 * j;
    float t = 0.f;
    for (int hh = 0; hh < heads; ++hh)
      t += accum[(size_t)node * heads * kHID + hh * kHID + c];
    t = t * inv_h + bias[c];
    v[j] = t; s1 += t; s2 += t * t;
  }
  for (int off = 16; off; off >>= 1) {
    s1 += __shfl_xor(s1, off, 32);
    s2 += __shfl_xor(s2, off, 32);
  }
  float mu   = s1 * (1.f / 128.f);
  float var  = s2 * (1.f / 128.f) - mu * mu;
  float rstd = rsqrtf(var + 1e-5f);
#pragma unroll
  for (int j = 0; j < 4; ++j) {
    int c = lane + 32 * j;
    float y = (v[j] - mu) * rstd * gam[c] + bet[c];
    y = y > 0.f ? y : __expf(y) - 1.f;          // elu
    hout[(size_t)node * kHID + c] = (_Float16)y;
  }
}

// fused per-edge classifier: relu(er@c1w+b) -> relu(@c2w+b) -> @c3w+b
__global__ void k_classifier(const _Float16* __restrict__ h2, const _Float16* __restrict__ ea16,
                             const int* __restrict__ src, const int* __restrict__ dst,
                             const _Float16* __restrict__ c1w_sw, const float* __restrict__ c1b,
                             const _Float16* __restrict__ c2w_sw, const float* __restrict__ c2b,
                             const float* __restrict__ c3w, const float* __restrict__ c3b,
                             float* __restrict__ out) {
  __shared__ _Float16 At[16 * kCIN];
  __shared__ _Float16 Ht[16 * kHID];
  __shared__ _Float16 H2t[16 * 64];
  __shared__ int sIdx[16], dIdx[16];
  int tid = threadIdx.x, tile = blockIdx.x;
  if (tid < 16) {
    int s = src[tile * 16 + tid], d = dst[tile * 16 + tid];
    sIdx[tid] = s; dIdx[tid] = d;
    __builtin_prefetch(h2 + (size_t)s * kHID, 0, 0);
    __builtin_prefetch(h2 + (size_t)d * kHID, 0, 0);
  }
  __syncthreads();
  for (int i = tid; i < 16 * kCIN; i += 256) {
    int m = i / kCIN, c = i % kCIN;
    _Float16 v;
    if (c < 128)      v = h2[(size_t)sIdx[m] * kHID + c];
    else if (c < 256) v = h2[(size_t)dIdx[m] * kHID + (c - 128)];
    else              v = ea16[(size_t)(tile * 16 + m) * kED + (c - 256)];
    At[i] = v;
  }
  __syncthreads();
  int wave = tid >> 5, lane = tid & 31, hf = lane >> 4;
  {  // stage 1: [16,320] @ [320,128]
    v8f acc = {0.f, 0.f, 0.f, 0.f, 0.f, 0.f, 0.f, 0.f};
#pragma unroll
    for (int k0 = 0; k0 < kCIN; k0 += 32)
      acc = WMMA_F16(false, wmma_load_a_lds(At, kCIN, k0, lane),
                     false, wmma_load_b_sw(c1w_sw, kHID, k0, wave * 16, lane),
                     (short)0, acc, false, false);
    int col = wave * 16 + (lane & 15);
    float bias = c1b[col];
#pragma unroll
    for (int r = 0; r < 8; ++r)
      Ht[(r + 8 * hf) * kHID + col] = (_Float16)fmaxf(acc[r] + bias, 0.f);
  }
  __syncthreads();
  if (wave < 4) {  // stage 2: [16,128] @ [128,64]
    v8f acc = {0.f, 0.f, 0.f, 0.f, 0.f, 0.f, 0.f, 0.f};
#pragma unroll
    for (int k0 = 0; k0 < kHID; k0 += 32)
      acc = WMMA_F16(false, wmma_load_a_lds(Ht, kHID, k0, lane),
                     false, wmma_load_b_sw(c2w_sw, 64, k0, wave * 16, lane),
                     (short)0, acc, false, false);
    int col = wave * 16 + (lane & 15);
    float bias = c2b[col];
#pragma unroll
    for (int r = 0; r < 8; ++r)
      H2t[(r + 8 * hf) * 64 + col] = (_Float16)fmaxf(acc[r] + bias, 0.f);
  }
  __syncthreads();
  if (tid < 16) {  // stage 3: [16,64] @ [64,1]
    float s = c3b[0];
    for (int k = 0; k < 64; ++k) s += (float)H2t[tid * 64 + k] * c3w[k];
    out[tile * 16 + tid] = s;
  }
}

// ---------------------------------------------------------------------------
// Host launch
// ---------------------------------------------------------------------------
extern "C" void kernel_launch(void* const* d_in, const int* in_sizes, int n_in,
                              void* d_out, int out_size, void* d_ws, size_t ws_size,
                              hipStream_t stream) {
  (void)in_sizes; (void)n_in; (void)out_size; (void)ws_size;
  const float* node_stats = (const float*)d_in[1];
  const int*   eidx       = (const int*)d_in[2];
  const int*   src = eidx, *dst = eidx + kE;
  const float* edge_attr = (const float*)d_in[3];
  const float* epw  = (const float*)d_in[4];
  const float* epb  = (const float*)d_in[5];
  const float* g1wl = (const float*)d_in[6];
  const float* g1wr = (const float*)d_in[7];
  const float* g1we = (const float*)d_in[8];
  const float* g1att= (const float*)d_in[9];
  const float* g1b  = (const float*)d_in[10];
  const float* n1g  = (const float*)d_in[11];
  const float* n1b  = (const float*)d_in[12];
  const float* g2wl = (const float*)d_in[13];
  const float* g2wr = (const float*)d_in[14];
  const float* g2we = (const float*)d_in[15];
  const float* g2att= (const float*)d_in[16];
  const float* g2b  = (const float*)d_in[17];
  const float* n2g  = (const float*)d_in[18];
  const float* n2b  = (const float*)d_in[19];
  const float* c1w  = (const float*)d_in[20];
  const float* c1b  = (const float*)d_in[21];
  const float* c2w  = (const float*)d_in[22];
  const float* c2b  = (const float*)d_in[23];
  const float* c3w  = (const float*)d_in[24];
  const float* c3b  = (const float*)d_in[25];
  float* outp = (float*)d_out;

  // --- workspace carve-up ---
  char* base = (char*)d_ws;
  size_t off = 0;
  auto alloc = [&](size_t bytes) { void* p = base + off; off = (off + bytes + 255) & ~(size_t)255; return p; };
  _Float16* ea16     = (_Float16*)alloc((size_t)kE * kED * 2);
  _Float16* xi16     = (_Float16*)alloc((size_t)kN * kGIN * 2);
  float*    loop_sum = (float*)   alloc((size_t)kN * kED * 4);
  _Float16* loop16   = (_Float16*)alloc((size_t)kN * kED * 2);
  float*    cnt_out  = (float*)   alloc((size_t)kN * 4);
  float*    cnt_in   = (float*)   alloc((size_t)kN * 4);
  float*    xsum_out = (float*)   alloc((size_t)kN * kND * 4);
  float*    xsum_in  = (float*)   alloc((size_t)kN * kND * 4);
  _Float16* xl1      = (_Float16*)alloc((size_t)kN * 256 * 2);
  _Float16* xr1      = (_Float16*)alloc((size_t)kN * 256 * 2);
  _Float16* we1      = (_Float16*)alloc((size_t)kE * 256 * 2);
  _Float16* lwe1     = (_Float16*)alloc((size_t)kN * 256 * 2);
  float*    logit1   = (float*)   alloc((size_t)kE2 * 2 * 4);
  unsigned* mkey1    = (unsigned*)alloc((size_t)kN * 2 * 4);
  float*    denom1   = (float*)   alloc((size_t)kN * 2 * 4);
  float*    out1     = (float*)   alloc((size_t)kN * 256 * 4);
  _Float16* h1       = (_Float16*)alloc((size_t)kN * kHID * 2);
  _Float16* xl2      = (_Float16*)alloc((size_t)kN * kHID * 2);
  _Float16* xr2      = (_Float16*)alloc((size_t)kN * kHID * 2);
  _Float16* we2      = (_Float16*)alloc((size_t)kE * kHID * 2);
  _Float16* lwe2     = (_Float16*)alloc((size_t)kN * kHID * 2);
  float*    logit2   = (float*)   alloc((size_t)kE2 * 4);
  unsigned* mkey2    = (unsigned*)alloc((size_t)kN * 4);
  float*    denom2   = (float*)   alloc((size_t)kN * 4);
  float*    out2     = (float*)   alloc((size_t)kN * kHID * 4);
  _Float16* h2       = (_Float16*)alloc((size_t)kN * kHID * 2);
  _Float16* w_epw  = (_Float16*)alloc((size_t)64 * 128 * 2);
  _Float16* w_g1wl = (_Float16*)alloc((size_t)288 * 256 * 2);
  _Float16* w_g1wr = (_Float16*)alloc((size_t)288 * 256 * 2);
  _Float16* w_g1we = (_Float16*)alloc((size_t)64 * 256 * 2);
  _Float16* w_g2wl = (_Float16*)alloc((size_t)128 * 128 * 2);
  _Float16* w_g2wr = (_Float16*)alloc((size_t)128 * 128 * 2);
  _Float16* w_g2we = (_Float16*)alloc((size_t)64 * 128 * 2);
  _Float16* w_c1w  = (_Float16*)alloc((size_t)320 * 128 * 2);
  _Float16* w_c2w  = (_Float16*)alloc((size_t)128 * 64 * 2);

  // --- zero accumulators (every call: harness does not re-poison) ---
  hipMemsetAsync(loop_sum, 0, (size_t)kN * kED * 4, stream);
  hipMemsetAsync(cnt_out,  0, (size_t)kN * 4, stream);
  hipMemsetAsync(cnt_in,   0, (size_t)kN * 4, stream);
  hipMemsetAsync(xsum_out, 0, (size_t)kN * kND * 4, stream);
  hipMemsetAsync(xsum_in,  0, (size_t)kN * kND * 4, stream);
  hipMemsetAsync(mkey1,    0, (size_t)kN * 2 * 4, stream);
  hipMemsetAsync(denom1,   0, (size_t)kN * 2 * 4, stream);
  hipMemsetAsync(out1,     0, (size_t)kN * 256 * 4, stream);
  hipMemsetAsync(mkey2,    0, (size_t)kN * 4, stream);
  hipMemsetAsync(denom2,   0, (size_t)kN * 4, stream);
  hipMemsetAsync(out2,     0, (size_t)kN * kHID * 4, stream);

  auto g1 = [](long total) { return dim3((unsigned)((total + 255) / 256)); };

  // --- weight swizzles (WMMA fragment order) + input f16 conversion ---
  k_swz_b<<<g1((long)64 * 128), 256, 0, stream>>>(epw, w_epw, 64, 128, 64L * 128);
  k_swz_b<<<g1((long)288 * 256), 256, 0, stream>>>(g1wl, w_g1wl, 258, 256, 288L * 256);
  k_swz_b<<<g1((long)288 * 256), 256, 0, stream>>>(g1wr, w_g1wr, 258, 256, 288L * 256);
  k_swz_b<<<g1((long)64 * 256), 256, 0, stream>>>(g1we, w_g1we, 64, 256, 64L * 256);
  k_swz_b<<<g1((long)128 * 128), 256, 0, stream>>>(g2wl, w_g2wl, 128, 128, 128L * 128);
  k_swz_b<<<g1((long)128 * 128), 256, 0, stream>>>(g2wr, w_g2wr, 128, 128, 128L * 128);
  k_swz_b<<<g1((long)64 * 128), 256, 0, stream>>>(g2we, w_g2we, 64, 128, 64L * 128);
  k_swz_b<<<g1((long)320 * 128), 256, 0, stream>>>(c1w, w_c1w, 320, 128, 320L * 128);
  k_swz_b<<<g1((long)128 * 64), 256, 0, stream>>>(c2w, w_c2w, 128, 64, 128L * 64);
  k_cvt<<<g1((long)kE * kED), 256, 0, stream>>>(edge_attr, ea16, kE, kED, (long)kE * kED);

  // --- stage 0: counts, self-loop attr, edge MLP + segment sums ---
  k_count<<<g1(kE), 256, 0, stream>>>(src, dst, cnt_out, cnt_in);
  k_loopsum<<<g1((long)kE * kED), 256, 0, stream>>>(edge_attr, dst, loop_sum);
  k_edge_emb<<<kE / 16, 256, 0, stream>>>(edge_attr, src, dst, w_epw, epb, xsum_out, xsum_in);
  k_build_xi<<<g1((long)kN * kGIN), 256, 0, stream>>>(xsum_out, xsum_in, cnt_out, cnt_in,
                                                      node_stats, xi16);
  k_finalize_loop<<<g1((long)kN * kED), 256, 0, stream>>>(loop_sum, cnt_in, loop16);

  // --- GATv2 layer 1 (heads=2, C=128) ---
  k_gemm16<<<dim3(kN / 16, 4), 128, 0, stream>>>(xi16, w_g1wl, xl1, kGIN, 256);
  k_gemm16<<<dim3(kN / 16, 4), 128, 0, stream>>>(xi16, w_g1wr, xr1, kGIN, 256);
  k_gemm16<<<dim3(kE / 16, 4), 128, 0, stream>>>(ea16, w_g1we, we1, kED, 256);
  k_gemm16<<<dim3(kN / 16, 4), 128, 0, stream>>>(loop16, w_g1we, lwe1, kED, 256);
  k_logit<<<g1((long)kE2 * 2), 256, 0, stream>>>(xl1, xr1, we1, lwe1, g1att, src, dst,
                                                 logit1, mkey1, 2);
  k_denom<<<g1((long)kE2 * 2), 256, 0, stream>>>(logit1, mkey1, denom1, dst, 2);
  k_attn_accum<<<g1((long)kE2 * 2), 256, 0, stream>>>(xl1, logit1, mkey1, denom1, src, dst,
                                                      out1, 2);
  k_node_norm<<<kN / 8, 256, 0, stream>>>(out1, 2, g1b, n1g, n1b, h1);

  // --- GATv2 layer 2 (heads=1, C=128) ---
  k_gemm16<<<dim3(kN / 16, 2), 128, 0, stream>>>(h1, w_g2wl, xl2, kHID, 128);
  k_gemm16<<<dim3(kN / 16, 2), 128, 0, stream>>>(h1, w_g2wr, xr2, kHID, 128);
  k_gemm16<<<dim3(kE / 16, 2), 128, 0, stream>>>(ea16, w_g2we, we2, kED, 128);
  k_gemm16<<<dim3(kN / 16, 2), 128, 0, stream>>>(loop16, w_g2we, lwe2, kED, 128);
  k_logit<<<g1((long)kE2), 256, 0, stream>>>(xl2, xr2, we2, lwe2, g2att, src, dst,
                                             logit2, mkey2, 1);
  k_denom<<<g1((long)kE2), 256, 0, stream>>>(logit2, mkey2, denom2, dst, 1);
  k_attn_accum<<<g1((long)kE2), 256, 0, stream>>>(xl2, logit2, mkey2, denom2, src, dst,
                                                  out2, 1);
  k_node_norm<<<kN / 8, 256, 0, stream>>>(out2, 1, g2b, n2g, n2b, h2);

  // --- fused per-edge classifier ---
  k_classifier<<<kE / 16, 256, 0, stream>>>(h2, ea16, src, dst, w_c1w, c1b, w_c2w, c2b,
                                            c3w, c3b, outp);
}
